// DFS_CL_Attention_17136919511589
// MI455X (gfx1250) — matrix operand order repaired and checked by reading, not compiled, not run on hardware
//
#include <hip/hip_runtime.h>

// ---------------------------------------------------------------------------
// DFS_CL_Attention for MI455X (gfx1250).
// Implicit-GEMM 3x3 convs on v_wmma_f32_16x16x32_f16 (f32 accum), weights
// pre-packed to f16 WMMA tiles and fed by the Tensor Data Mover
// (tensor_load_to_lds + s_wait_tensorcnt), channel-attention folded into LDS
// staging of the consumer conv, deformable conv = bilinear gather + WMMA.
// ---------------------------------------------------------------------------

typedef __attribute__((ext_vector_type(16))) _Float16     v16h;
typedef __attribute__((ext_vector_type(8)))  _Float16     v8h;
typedef __attribute__((ext_vector_type(8)))  float        v8f;
typedef __attribute__((ext_vector_type(4)))  unsigned int u32x4;
typedef __attribute__((ext_vector_type(4)))  int          i32x4;
typedef __attribute__((ext_vector_type(8)))  int          i32x8;

#define HH 192
#define WW 192
#define NB 8
#define PIX_T 32
#define WPB (WW / PIX_T)

#define MODE_PLAIN   0
#define MODE_CONCAT  1
#define MODE_FLIPCAT 2

#define A_STRIDE 40          // halves per LDS row (80B, 16B aligned everywhere)
#define B_STRIDE 40
#define TILE_HALF 2048       // one packed weight tile = 64 x 32 f16

#ifndef USE_TDM
#define USE_TDM 1
#endif

// ---------------------------------------------------------------------------
// TDM: async-copy one 4KB weight tile (2048 f16) global -> LDS, inserting
// 16B of LDS padding after every 64B so rows land at stride A_STRIDE.
// D# group0: count=1 | lds_addr | global_addr(57b) | type=2.
// D# group1: data_size=2B, pad_enable, pad_interval=16DW, pad_amount=4DW,
//            tensor_dim0=tile_dim0=2048, tile_dim1=1, dim0_stride=2048.
// ---------------------------------------------------------------------------
__device__ __forceinline__ void tdm_load_tile(unsigned lds_addr,
                                              const _Float16* __restrict__ gsrc) {
  const unsigned long long ga = (unsigned long long)(size_t)(const void*)gsrc;
  u32x4 g0;
  g0.x = 1u;                                                  // count=1 (user)
  g0.y = lds_addr;
  g0.z = (unsigned)(ga & 0xffffffffull);
  g0.w = (unsigned)((ga >> 32) & 0x01ffffffull) | (2u << 30); // type=2 ("image")
  i32x8 g1;
  g1[0] = (int)((1u << 16) | (1u << 20) | (3u << 22) | (3u << 25));
  g1[1] = (int)(2048u << 16);   // tensor_dim0[15:0] at bits 63:48
  g1[2] = (int)(1u << 16);      // tensor_dim0 hi = 0 | tensor_dim1 = 1
  g1[3] = (int)(2048u << 16);   // tensor_dim1 hi = 0 | tile_dim0 = 2048
  g1[4] = 1;                    // tile_dim1 = 1, tile_dim2 = 0
  g1[5] = 2048;                 // tensor_dim0_stride lo32
  g1[6] = 0;
  g1[7] = 0;
  i32x4 z4 = {0, 0, 0, 0};
#if defined(__clang_major__) && (__clang_major__ >= 23)
  i32x8 z8 = {0, 0, 0, 0, 0, 0, 0, 0};
  __builtin_amdgcn_tensor_load_to_lds(g0, g1, z4, z4, z8, 0);
#else
  __builtin_amdgcn_tensor_load_to_lds(g0, g1, z4, z4, 0);
#endif
}

__device__ __forceinline__ v16h frag_cat(v8h a, v8h b) {
  union { v16h v; v8h h[2]; } u;
  u.h[0] = a; u.h[1] = b;
  return u.v;
}

// ---------------------------------------------------------------------------
// Pre-pack conv weights f32 OIHW -> f16 WMMA tiles [9][CHUNKS][64][32],
// zero-padded in both m (Cout<64) and k (Cin not multiple of 32).
// ---------------------------------------------------------------------------
__global__ __launch_bounds__(256)
void pack_weights(const float* __restrict__ w, _Float16* __restrict__ out,
                  int Cin, int Cout)
{
  const int CH = (Cin + 31) >> 5;
  const int total = 9 * CH * 2048;
  for (int i = blockIdx.x * 256 + threadIdx.x; i < total; i += gridDim.x * 256) {
    const int k   = i & 31;
    const int m   = (i >> 5) & 63;
    const int t2  = i >> 11;
    const int ch  = t2 % CH;
    const int tap = t2 / CH;
    const int c   = ch * 32 + k;
    float v = 0.f;
    if (m < Cout && c < Cin)
      v = w[(((size_t)m * Cin + c) * 3 + (tap / 3)) * 3 + (tap % 3)];
    out[i] = (_Float16)v;
  }
}

// ---------------------------------------------------------------------------
// Generic 3x3 conv, stride 1, pad 1, implicit GEMM.
// Block: 256 threads (8 waves, 4x2), tile 64 Cout x 32 pixels.
// ---------------------------------------------------------------------------
template<int CIN, int COUT, int MODE, bool HAS_SCALE, bool RELU, bool DUP>
__global__ __launch_bounds__(256)
void wmma_conv3x3(const float* __restrict__ src0,
                  const float* __restrict__ src1,    // MODE_CONCAT second half
                  const float* __restrict__ scale,   // [N*CIN] CA scale
                  const _Float16* __restrict__ pw,   // packed [9][CHUNKS][64][32]
                  const float* __restrict__ bias,
                  float* __restrict__ dst,
                  float* __restrict__ dst2)
{
  constexpr int CHUNKS = (CIN + 31) / 32;
  __shared__ alignas(16) _Float16 As[64 * A_STRIDE];
  __shared__ alignas(16) _Float16 Bs[32 * B_STRIDE];

  const int tid = threadIdx.x;
  const int bx  = blockIdx.x;
  const int oxb = (bx % WPB) * PIX_T;
  const int oy  = (bx / WPB) % HH;
  const int n   = bx / (WPB * HH);

  const int lane = tid & 31;
  const int wv   = tid >> 5;
  const int wm   = wv >> 1;
  const int wn   = wv & 1;
  const int half = lane >> 4;
  const int lr   = lane & 15;

  v8f acc = {};

  for (int tap = 0; tap < 9; ++tap) {
    const int ky = tap / 3, kx = tap % 3;
    const int iy = oy + ky - 1;
    #pragma unroll
    for (int ch = 0; ch < CHUNKS; ++ch) {
      // ---- A tile: packed f16 weights, contiguous 4KB -> LDS
#if USE_TDM
      if (wv == 0)
        tdm_load_tile((unsigned)(size_t)(void*)As,
                      pw + (size_t)(tap * CHUNKS + ch) * TILE_HALF);
#else
      {
        const _Float16* g = pw + (size_t)(tap * CHUNKS + ch) * TILE_HALF + tid * 8;
        *(v8h*)(As + (tid >> 2) * A_STRIDE + (tid & 3) * 8) = *(const v8h*)g;
      }
#endif
      // ---- B tile: im2col patch, f32 -> f16, transposed [pixel][k]
      #pragma unroll
      for (int q = 0; q < 4; ++q) {
        const int e = tid + q * 256;
        const int k = e >> 5, p = e & 31;
        const int c = ch * 32 + k;
        const int ix = oxb + p + kx - 1;
        const bool inb = ((unsigned)iy < (unsigned)HH) & ((unsigned)ix < (unsigned)WW);
        float v = 0.f;
        if constexpr (MODE == MODE_PLAIN) {
          if (inb) {
            v = src0[(((size_t)n * CIN + c) * HH + iy) * WW + ix];
            if constexpr (HAS_SCALE) v *= scale[n * CIN + c];
          }
        } else if constexpr (MODE == MODE_CONCAT) {
          if (inb) {
            v = (c < 64)
              ? src0[(((size_t)n * 64 + c)        * HH + iy) * WW + ix]
              : src1[(((size_t)n * 64 + (c - 64)) * HH + iy) * WW + ix];
          }
        } else { // MODE_FLIPCAT: 12ch from 3ch via {id, flipH, flipW, flipHW}
          if (inb && c < CIN) {
            const int g  = c / 3, cc = c - g * 3;
            const int yy = (g & 1) ? (HH - 1 - iy) : iy;
            const int xx = (g & 2) ? (WW - 1 - ix) : ix;
            v = src0[(((size_t)n * 3 + cc) * HH + yy) * WW + xx];
          }
        }
        Bs[p * B_STRIDE + k] = (_Float16)v;
      }
#if USE_TDM
      if (wv == 0) __builtin_amdgcn_s_wait_tensorcnt(0);
#endif
      __syncthreads();

      // ---- fragments: two contiguous b128 LDS loads each
      const _Float16* arow = As + (wm * 16 + lr) * A_STRIDE;
      const _Float16* brow = Bs + (wn * 16 + lr) * B_STRIDE;
      const v16h af = frag_cat(*(const v8h*)(arow + half * 8),
                               *(const v8h*)(arow + 16 + half * 8));
      const v16h bf = frag_cat(*(const v8h*)(brow + half * 8),
                               *(const v8h*)(brow + 16 + half * 8));
      acc = __builtin_amdgcn_wmma_f32_16x16x32_f16(
          false, af, false, bf, (short)0, acc, false, false);
      __syncthreads();
    }
  }

  #pragma unroll
  for (int v = 0; v < 8; ++v) {
    const int m = wm * 16 + v + half * 8;
    const int p = wn * 16 + lr;
    if (m < COUT) {
      float r = acc[v] + bias[m];
      if constexpr (RELU) r = fmaxf(r, 0.f);
      const size_t o = (((size_t)n * COUT + m) * HH + oy) * WW + (oxb + p);
      dst[o] = r;
      if constexpr (DUP) dst2[o] = r;
    }
  }
}

// ---------------------------------------------------------------------------
// Deformable 3x3 conv: per tap, 32 lanes compute sample coords; 256 threads
// bilinear-gather the 64ch x 32px tile to LDS f16; WMMA over 9 taps x 2 chunks.
// ---------------------------------------------------------------------------
__global__ __launch_bounds__(256)
void wmma_deform_conv(const float* __restrict__ x,
                      const float* __restrict__ off,     // [N][18][H][W]
                      const _Float16* __restrict__ pw,   // packed [9][2][64][32]
                      const float* __restrict__ bias,
                      float* __restrict__ xo)
{
  __shared__ alignas(16) _Float16 As[64 * A_STRIDE];
  __shared__ alignas(16) _Float16 Bs[32 * B_STRIDE];
  __shared__ float spy[PIX_T], spx[PIX_T];

  const int tid = threadIdx.x;
  const int bx  = blockIdx.x;
  const int oxb = (bx % WPB) * PIX_T;
  const int oy  = (bx / WPB) % HH;
  const int n   = bx / (WPB * HH);

  const int lane = tid & 31;
  const int wv   = tid >> 5;
  const int wm   = wv >> 1;
  const int wn   = wv & 1;
  const int half = lane >> 4;
  const int lr   = lane & 15;

  v8f acc = {};

  for (int tap = 0; tap < 9; ++tap) {
    const int i = tap / 3, j = tap % 3;

    if (tid < PIX_T) {
      const int p = tid;
      const float dy = off[(((size_t)n * 18 + 2 * tap    ) * HH + oy) * WW + (oxb + p)];
      const float dx = off[(((size_t)n * 18 + 2 * tap + 1) * HH + oy) * WW + (oxb + p)];
      spy[p] = (float)(oy - 1 + i) + dy;
      spx[p] = (float)(oxb + p - 1 + j) + dx;
    }
    __syncthreads();

    #pragma unroll
    for (int ch = 0; ch < 2; ++ch) {
#if USE_TDM
      if (wv == 0)
        tdm_load_tile((unsigned)(size_t)(void*)As,
                      pw + (size_t)(tap * 2 + ch) * TILE_HALF);
#else
      {
        const _Float16* g = pw + (size_t)(tap * 2 + ch) * TILE_HALF + tid * 8;
        *(v8h*)(As + (tid >> 2) * A_STRIDE + (tid & 3) * 8) = *(const v8h*)g;
      }
#endif
      // ---- B: bilinear gather 32ch x 32px
      #pragma unroll
      for (int q = 0; q < 4; ++q) {
        const int e = tid + q * 256;
        const int k = e >> 5, p = e & 31;
        const int c = ch * 32 + k;
        const float py = spy[p], px = spx[p];
        const float fy = floorf(py), fx = floorf(px);
        const int   y0 = (int)fy,   x0 = (int)fx;
        const float wy1 = py - fy,  wx1 = px - fx;
        const float wy0 = 1.f - wy1, wx0 = 1.f - wx1;
        const float* base = x + ((size_t)n * 64 + c) * (size_t)(HH * WW);
        auto ld = [&](int y, int xx) -> float {
          return ((unsigned)y < (unsigned)HH && (unsigned)xx < (unsigned)WW)
               ? base[y * WW + xx] : 0.f;
        };
        const float v = ld(y0,     x0    ) * (wy0 * wx0)
                      + ld(y0,     x0 + 1) * (wy0 * wx1)
                      + ld(y0 + 1, x0    ) * (wy1 * wx0)
                      + ld(y0 + 1, x0 + 1) * (wy1 * wx1);
        Bs[p * B_STRIDE + k] = (_Float16)v;
      }
#if USE_TDM
      if (wv == 0) __builtin_amdgcn_s_wait_tensorcnt(0);
#endif
      __syncthreads();

      const _Float16* arow = As + (wm * 16 + lr) * A_STRIDE;
      const _Float16* brow = Bs + (wn * 16 + lr) * B_STRIDE;
      const v16h af = frag_cat(*(const v8h*)(arow + half * 8),
                               *(const v8h*)(arow + 16 + half * 8));
      const v16h bf = frag_cat(*(const v8h*)(brow + half * 8),
                               *(const v8h*)(brow + 16 + half * 8));
      acc = __builtin_amdgcn_wmma_f32_16x16x32_f16(
          false, af, false, bf, (short)0, acc, false, false);
      __syncthreads();
    }
  }

  #pragma unroll
  for (int v = 0; v < 8; ++v) {
    const int m = wm * 16 + v + half * 8;
    const int p = wn * 16 + lr;
    const float r = fmaxf(acc[v] + bias[m], 0.f);
    xo[(((size_t)n * 64 + m) * HH + oy) * WW + (oxb + p)] = r;
  }
}

// ---------------------------------------------------------------------------
__global__ __launch_bounds__(256)
void gap_kernel(const float* __restrict__ src, float* __restrict__ gap)
{
  const int nc = blockIdx.x;
  const float* p = src + (size_t)nc * (HH * WW);
  float s = 0.f;
  for (int i = threadIdx.x; i < HH * WW; i += 256) {
    if (i + 4096 < HH * WW) __builtin_prefetch(p + i + 4096, 0, 3);
    s += p[i];
  }
  __shared__ float red[256];
  red[threadIdx.x] = s;
  __syncthreads();
  for (int st = 128; st > 0; st >>= 1) {
    if (threadIdx.x < st) red[threadIdx.x] += red[threadIdx.x + st];
    __syncthreads();
  }
  if (threadIdx.x == 0) gap[nc] = red[0] * (1.f / (HH * WW));
}

__global__ __launch_bounds__(64)
void ca_scale_kernel(const float* __restrict__ gap,
                     const float* __restrict__ w1, const float* __restrict__ b1,
                     const float* __restrict__ w2, const float* __restrict__ b2,
                     float* __restrict__ sc)
{
  const int n = blockIdx.x;
  const int t = threadIdx.x;
  __shared__ float hid[4];
  if (t < 4) {
    float a = b1[t];
    for (int c = 0; c < 64; ++c) a += gap[n * 64 + c] * w1[t * 64 + c];
    hid[t] = fmaxf(a, 0.f);
  }
  __syncthreads();
  float a = b2[t];
  #pragma unroll
  for (int q = 0; q < 4; ++q) a += hid[q] * w2[t * 4 + q];
  sc[n * 64 + t] = 1.f / (1.f + __expf(-a));
}

// ---------------------------------------------------------------------------
extern "C" void kernel_launch(void* const* d_in, const int* in_sizes, int n_in,
                              void* d_out, int out_size, void* d_ws, size_t ws_size,
                              hipStream_t stream) {
  (void)in_sizes; (void)n_in; (void)out_size; (void)ws_size;

  const float* x      = (const float*)d_in[0];
  const float* ref_x  = (const float*)d_in[1];
  const float* ref_w  = (const float*)d_in[2];
  const float* ref_b  = (const float*)d_in[3];
  const float* off_w1 = (const float*)d_in[4];
  const float* off_b1 = (const float*)d_in[5];
  const float* ca1_w1 = (const float*)d_in[6];
  const float* ca1_b1 = (const float*)d_in[7];
  const float* ca1_w2 = (const float*)d_in[8];
  const float* ca1_b2 = (const float*)d_in[9];
  const float* off_w2 = (const float*)d_in[10];
  const float* off_b2 = (const float*)d_in[11];
  const float* ca2_w1 = (const float*)d_in[12];
  const float* ca2_b1 = (const float*)d_in[13];
  const float* ca2_w2 = (const float*)d_in[14];
  const float* ca2_b2 = (const float*)d_in[15];
  const float* off_w3 = (const float*)d_in[16];
  const float* off_b3 = (const float*)d_in[17];
  const float* dcn_w  = (const float*)d_in[18];
  const float* dcn_b  = (const float*)d_in[19];
  const float* rgb_w  = (const float*)d_in[20];
  const float* rgb_b  = (const float*)d_in[21];

  float* out = (float*)d_out;
  const size_t FEAT = (size_t)NB * 64 * HH * WW;
  const size_t RGBN = (size_t)NB * 3  * HH * WW;
  const size_t OFFN = (size_t)NB * 18 * HH * WW;
  float* out_xo   = out;
  float* out_rgb  = out + FEAT;
  float* out_mask = out + FEAT + RGBN;
  float* out_off  = out + FEAT + RGBN + OFFN;

  float* bufA = (float*)d_ws;          // ref_f, then t2
  float* bufB = bufA + FEAT;           // t1
  float* gapb = bufB + FEAT;
  float* s1   = gapb + 512;
  float* s2   = s1 + 512;

  // packed f16 weight tiles (256B aligned)
  size_t hb = (size_t)(s2 + 512);
  hb = (hb + 255) & ~(size_t)255;
  _Float16* pw_ref = (_Float16*)hb;            // 9*1*2048
  _Float16* pw_w1  = pw_ref + 9 * 1 * 2048;    // 9*4*2048
  _Float16* pw_w2  = pw_w1  + 9 * 4 * 2048;    // 9*2*2048
  _Float16* pw_w3  = pw_w2  + 9 * 2 * 2048;
  _Float16* pw_dcn = pw_w3  + 9 * 2 * 2048;
  _Float16* pw_rgb = pw_dcn + 9 * 2 * 2048;

  const dim3 blk(256);
  const dim3 grd(WPB * HH * NB, 1);
  const dim3 pgrd(128);

  // 0) pack all conv weights into WMMA-ready f16 tiles (L2-resident)
  pack_weights<<<pgrd, blk, 0, stream>>>(ref_w,  pw_ref, 12,  64);
  pack_weights<<<pgrd, blk, 0, stream>>>(off_w1, pw_w1,  128, 64);
  pack_weights<<<pgrd, blk, 0, stream>>>(off_w2, pw_w2,  64,  64);
  pack_weights<<<pgrd, blk, 0, stream>>>(off_w3, pw_w3,  64,  18);
  pack_weights<<<pgrd, blk, 0, stream>>>(dcn_w,  pw_dcn, 64,  64);
  pack_weights<<<pgrd, blk, 0, stream>>>(rgb_w,  pw_rgb, 64,  3);

  // 1) ref branch: flip-concat (12ch) -> conv -> relu -> ref_f (bufA)
  wmma_conv3x3<12, 64, MODE_FLIPCAT, false, true, false>
      <<<grd, blk, 0, stream>>>(ref_x, nullptr, nullptr, pw_ref, ref_b, bufA, nullptr);
  // 2) offset conv1: concat(x, ref_f) -> relu -> t1 (bufB)
  wmma_conv3x3<128, 64, MODE_CONCAT, false, true, false>
      <<<grd, blk, 0, stream>>>(x, bufA, nullptr, pw_w1, off_b1, bufB, nullptr);
  // 3) CA1
  gap_kernel<<<dim3(NB * 64), blk, 0, stream>>>(bufB, gapb);
  ca_scale_kernel<<<dim3(NB), dim3(64), 0, stream>>>(gapb, ca1_w1, ca1_b1, ca1_w2, ca1_b2, s1);
  // 4) offset conv2 (t1*s1 folded) -> relu -> t2 (bufA)
  wmma_conv3x3<64, 64, MODE_PLAIN, true, true, false>
      <<<grd, blk, 0, stream>>>(bufB, nullptr, s1, pw_w2, off_b2, bufA, nullptr);
  // 5) CA2
  gap_kernel<<<dim3(NB * 64), blk, 0, stream>>>(bufA, gapb);
  ca_scale_kernel<<<dim3(NB), dim3(64), 0, stream>>>(gapb, ca2_w1, ca2_b1, ca2_w2, ca2_b2, s2);
  // 6) offset conv3 (t2*s2 folded) -> offset + mask (no relu)
  wmma_conv3x3<64, 18, MODE_PLAIN, true, false, true>
      <<<grd, blk, 0, stream>>>(bufA, nullptr, s2, pw_w3, off_b3, out_off, out_mask);
  // 7) deformable conv -> relu -> xo
  wmma_deform_conv<<<grd, blk, 0, stream>>>(x, out_off, pw_dcn, dcn_b, out_xo);
  // 8) rgb conv (no relu)
  wmma_conv3x3<64, 3, MODE_PLAIN, false, false, false>
      <<<grd, blk, 0, stream>>>(out_xo, nullptr, nullptr, pw_rgb, rgb_b, out_rgb, nullptr);
}
